// SynthesisBlock_51384988729654
// MI455X (gfx1250) — matrix operand order repaired and checked by reading, not compiled, not run on hardware
//
#include <hip/hip_runtime.h>
#include <hip/hip_bf16.h>

typedef __bf16  v16bf __attribute__((ext_vector_type(16)));
typedef float   v8f   __attribute__((ext_vector_type(8)));

#define BATCH   4
#define CIN     64
#define COUT    64
#define HH      512
#define WW      512
#define WDIM    512
#define KTAPS   9
#define CHPAD   72            // padded channel stride in LDS (bf16 elems)
#define TILE    8             // output pixels per block edge
#define HALO    10            // TILE + 2
#define KCHUNKS 18            // 576 / 32

struct alignas(32) Frag2 { uint4 lo, hi; };

static __device__ __forceinline__ unsigned short f2bf(float f) {
    union { float f; unsigned u; } v; v.f = f;
    unsigned u = v.u;
    unsigned r = u + 0x7FFFu + ((u >> 16) & 1u);   // round-to-nearest-even
    return (unsigned short)(r >> 16);
}

// ---------------------------------------------------------------- kernel 1
// styles[b][c] = dot(ws[b,:], affine_weight[c,:]) / sqrt(512) + affine_bias[c]
__global__ void styles_kernel(const float* __restrict__ ws,
                              const float* __restrict__ aw,
                              const float* __restrict__ ab,
                              float* __restrict__ styles) {
    int b = blockIdx.x;
    int c = threadIdx.x;
    const float* wrow = ws + b * WDIM;
    const float* arow = aw + c * WDIM;
    float s = 0.f;
    #pragma unroll 4
    for (int i = 0; i < WDIM; ++i) s += wrow[i] * arow[i];
    styles[b * CIN + c] = s * 0.044194173824159216f /* 1/sqrt(512) */ + ab[c];
}

// ---------------------------------------------------------------- kernel 2
// Modulate + demodulate, emit bf16 weights pre-swizzled into WMMA A-fragment
// layout: aprep[(((b*4+mt)*18+kc)*32 + lane)*16 + p].
// K ordering: k = tap*64 + c_in  (tap = kh*3+kw).
__global__ void wprep_kernel(const float* __restrict__ cw,
                             const float* __restrict__ styles,
                             unsigned short* __restrict__ aprep) {
    int bc = blockIdx.x;            // b*COUT + c_out
    int b = bc >> 6;
    int c_out = bc & 63;
    int cin = threadIdx.x;          // 0..63

    float s = styles[b * CIN + cin];
    const float* wp = cw + (c_out * CIN + cin) * KTAPS;
    float wv[KTAPS];
    float ssum = 0.f;
    #pragma unroll
    for (int t = 0; t < KTAPS; ++t) { wv[t] = wp[t] * s; ssum += wv[t] * wv[t]; }

    __shared__ float red[CIN];
    red[cin] = ssum;
    __syncthreads();
    for (int st = 32; st > 0; st >>= 1) {
        if (cin < st) red[cin] += red[cin + st];
        __syncthreads();
    }
    float demod = rsqrtf(red[0] + 1e-8f);

    int mt = c_out >> 4;
    int m  = c_out & 15;
    int hc = cin >> 5;              // which 32-channel half of a tap
    int j  = cin & 31;              // position within K=32 chunk
    int lh = j >> 4;                // lane half (contiguous k = 16*h + s)
    int p  = j & 15;
    int lane = lh * 16 + m;
    #pragma unroll
    for (int t = 0; t < KTAPS; ++t) {
        int kc = t * 2 + hc;
        size_t idx = ((((size_t)b * 4 + mt) * KCHUNKS + kc) * 32 + lane) * 16 + p;
        aprep[idx] = f2bf(wv[t] * demod);
    }
}

// ---------------------------------------------------------------- kernel 3
// Implicit-GEMM modulated conv + noise + bias + lrelu*sqrt(2) + clamp.
// Block: 256 thr (8 waves) -> 64 c_out x 8x8 pixels.  Wave w: mt = w>>1,
// two N-tiles (16 px = 2 rows x 8 cols each).
__global__ void __launch_bounds__(256)
conv_kernel(const float* __restrict__ x,
            const float* __restrict__ noise,
            const float* __restrict__ bias,
            const unsigned short* __restrict__ aprep,
            float* __restrict__ out) {
    __shared__ __align__(16) unsigned short smem[HALO * HALO * CHPAD];

    const int bx = blockIdx.x, by = blockIdx.y, b = blockIdx.z;
    const int tid = threadIdx.x;

    // ---- stage 10x10x64 fp32 halo tile -> bf16 LDS [pixel][chan(pad 72)]
    const int gy0 = by * TILE - 1;
    const int gx0 = bx * TILE - 1;
    const float* xb = x + (size_t)b * CIN * HH * WW;
    for (int e = tid; e < CIN * HALO * HALO; e += 256) {
        int c   = e / (HALO * HALO);
        int pix = e - c * (HALO * HALO);
        int py  = pix / HALO;
        int px  = pix - py * HALO;
        int gy = gy0 + py, gx = gx0 + px;
        float v = 0.f;
        if ((unsigned)gy < HH && (unsigned)gx < WW)
            v = xb[((size_t)c * HH + gy) * WW + gx];
        smem[pix * CHPAD + c] = f2bf(v);
    }
    __syncthreads();

    // ---- WMMA main loop (uniform control flow: EXEC all ones)
    const int wv   = tid >> 5;
    const int lane = tid & 31;
    const int mt   = wv >> 1;
    const int nt0  = (wv & 1) * 2;          // N-tile pair {nt0, nt0+1}

    const int n     = lane & 15;
    const int khalf = lane >> 4;
    const int prow  = n >> 3;               // row within N-tile (0/1)
    const int pcol  = n & 7;

    v8f acc0 = {0.f,0.f,0.f,0.f,0.f,0.f,0.f,0.f};
    v8f acc1 = {0.f,0.f,0.f,0.f,0.f,0.f,0.f,0.f};

    const unsigned short* abase =
        aprep + (((size_t)b * 4 + mt) * KCHUNKS) * 32 * 16 + lane * 16;

    #pragma unroll
    for (int kc = 0; kc < KCHUNKS; ++kc) {
        const int tap   = kc >> 1;
        const int chalf = kc & 1;
        const int dy = tap / 3, dx = tap - dy * 3;

        // A fragment: two global_load_b128 (L2-resident, pre-swizzled)
        const uint4* ap = (const uint4*)(abase + (size_t)kc * 512);
        Frag2 fa; fa.lo = ap[0]; fa.hi = ap[1];
        v16bf a = __builtin_bit_cast(v16bf, fa);

        // B fragments from LDS (im2col = pixel offset), two ds_load_b128 each
        const int px = pcol + dx;
        const int py0 = nt0 * 2 + prow + dy;
        const int off0 = (py0 * HALO + px) * CHPAD + chalf * 32 + khalf * 16;
        const uint4* bp0 = (const uint4*)(smem + off0);
        Frag2 fb0; fb0.lo = bp0[0]; fb0.hi = bp0[1];
        v16bf bf0 = __builtin_bit_cast(v16bf, fb0);

        const uint4* bp1 = (const uint4*)(smem + off0 + 2 * HALO * CHPAD);
        Frag2 fb1; fb1.lo = bp1[0]; fb1.hi = bp1[1];
        v16bf bf1 = __builtin_bit_cast(v16bf, fb1);

        acc0 = __builtin_amdgcn_wmma_f32_16x16x32_bf16(
                   false, a, false, bf0, (short)0, acc0, false, false);
        acc1 = __builtin_amdgcn_wmma_f32_16x16x32_bf16(
                   false, a, false, bf1, (short)0, acc1, false, false);
    }

    // ---- fused epilogue: noise + bias + lrelu*sqrt(2) + clamp
    const int gx  = bx * TILE + pcol;
    const int gy_0 = by * TILE + nt0 * 2 + prow;
    const int gy_1 = gy_0 + 2;
    const float nz0 = noise[((size_t)b * HH + gy_0) * WW + gx];
    const float nz1 = noise[((size_t)b * HH + gy_1) * WW + gx];

    #pragma unroll
    for (int r = 0; r < 8; ++r) {
        const int c_out = mt * 16 + r + 8 * khalf;
        const float bsv = bias[c_out];
        float* orow = out + ((size_t)(b * COUT + c_out) * HH) * WW + gx;

        float v0 = acc0[r] + nz0 + bsv;
        v0 = (v0 >= 0.f ? v0 : 0.2f * v0) * 1.4142135623730951f;
        v0 = fminf(fmaxf(v0, -256.f), 256.f);
        orow[(size_t)gy_0 * WW] = v0;

        float v1 = acc1[r] + nz1 + bsv;
        v1 = (v1 >= 0.f ? v1 : 0.2f * v1) * 1.4142135623730951f;
        v1 = fminf(fmaxf(v1, -256.f), 256.f);
        orow[(size_t)gy_1 * WW] = v1;
    }
}

// ----------------------------------------------------------------
extern "C" void kernel_launch(void* const* d_in, const int* in_sizes, int n_in,
                              void* d_out, int out_size, void* d_ws, size_t ws_size,
                              hipStream_t stream) {
    const float* x      = (const float*)d_in[0];
    const float* ws     = (const float*)d_in[1];
    const float* noise  = (const float*)d_in[2];
    const float* aw     = (const float*)d_in[3];
    const float* ab     = (const float*)d_in[4];
    const float* cw     = (const float*)d_in[5];
    const float* bias   = (const float*)d_in[6];
    float* out = (float*)d_out;

    float* styles = (float*)d_ws;                                  // 1 KB
    unsigned short* aprep = (unsigned short*)((char*)d_ws + 1024); // ~288 KB

    styles_kernel<<<dim3(BATCH), dim3(CIN), 0, stream>>>(ws, aw, ab, styles);
    wprep_kernel<<<dim3(BATCH * COUT), dim3(CIN), 0, stream>>>(cw, styles, aprep);
    conv_kernel<<<dim3(WW / TILE, HH / TILE, BATCH), dim3(256), 0, stream>>>(
        x, noise, bias, aprep, out);
}
